// Attention_55327768707775
// MI455X (gfx1250) — matrix operand order repaired and checked by reading, not compiled, not run on hardware
//
#include <hip/hip_runtime.h>
#include <hip/hip_bf16.h>

// ---------------------------------------------------------------------------
// GPT-2 attention block for gfx1250 (MI455X).
// bf16 WMMA + fp32 accumulate; ALL tile staging via async global->LDS DMA
// (ASYNCcnt, double-buffered ping-pong); global_prefetch for next stages.
//   NX=1024, H=16, D=64, S=2048, B=2
// Workspace layout (48 MiB total):
//   [0,   8M)  x_bf16          [4096 x 1024]   row-major
//   [8M, 14M)  w_attn^T bf16   [3072 x 1024]   (transposed: [n][k])
//   [14M,16M)  w_proj^T bf16   [1024 x 1024]   (transposed: [n][k])
//   [16M,24M)  Q bf16          [B*H, S, 64]    (pre-scaled by 1/8)
//   [24M,32M)  K bf16          [B*H, S, 64]
//   [32M,40M)  V^T bf16        [B*H, 64, S]    (d-major for verbatim DMA)
//   [40M,48M)  A bf16          [4096 x 1024]   (merged heads)
// ---------------------------------------------------------------------------

typedef __attribute__((ext_vector_type(16))) __bf16 v16bf;
typedef __attribute__((ext_vector_type(8)))  float  v8f;

union BfFrag {
    v16bf          v;
    unsigned short u16[16];
    unsigned int   u32[8];
};

__device__ __forceinline__ unsigned short f2bf(float f) {
    unsigned int u = __float_as_uint(f);
    u += 0x7fffu + ((u >> 16) & 1u);   // round to nearest even
    return (unsigned short)(u >> 16);
}

// gfx1250 async DMA: global -> LDS, 16 bytes per lane, tracked by ASYNCcnt.
__device__ __forceinline__ void async_g2l_b128(void* lds_ptr, const void* gptr) {
    unsigned lds_addr = (unsigned)(uintptr_t)lds_ptr;
    unsigned long long gaddr = (unsigned long long)(uintptr_t)gptr;
    asm volatile("global_load_async_to_lds_b128 %0, %1, off"
                 :: "v"(lds_addr), "v"(gaddr)
                 : "memory");
}
__device__ __forceinline__ void wait_async_le0() {
    asm volatile("s_wait_asynccnt 0" ::: "memory");
}
__device__ __forceinline__ void wait_async_le4() {
    asm volatile("s_wait_asynccnt 4" ::: "memory");
}
__device__ __forceinline__ void wait_async_le8() {
    asm volatile("s_wait_asynccnt 8" ::: "memory");
}

__global__ __launch_bounds__(256) void cvt_f32_to_bf16_kernel(
    const float* __restrict__ in, unsigned short* __restrict__ out, int n) {
    int i = blockIdx.x * 256 + threadIdx.x;
    if (i < n) out[i] = f2bf(in[i]);
}

// fp32 [K][N] -> bf16 transposed [N][K]  (one-time weight conversion)
__global__ __launch_bounds__(256) void cvt_f32_to_bf16_T_kernel(
    const float* __restrict__ in, unsigned short* __restrict__ out,
    int Kdim, int Ndim) {
    int i = blockIdx.x * 256 + threadIdx.x;
    if (i < Kdim * Ndim) {
        int k = i / Ndim;
        int n = i - k * Ndim;
        out[(size_t)n * Kdim + k] = f2bf(in[i]);
    }
}

// ---------------------------------------------------------------------------
// GEMM: C[M,N] = A[M,K](bf16 row-major) * B^T[N,K](bf16) + bias
// MODE 0: scatter into q/k/vT head-major buffers (bf16), scale q by 0.125
// MODE 1: plain fp32 output
// Block: 256 threads (8 waves), tile 128(M) x 128(N), K step 64,
// double-buffered pure-async LDS staging (8 b128 DMA calls / thread / stage).
// Waves: 4(M) x 2(N); each wave -> 32x64 via 2x4 frags x 2 k-chunks
// = 16 WMMA 16x16x32 per stage.
// ---------------------------------------------------------------------------
#define TILE_M 128
#define TILE_N 128
#define KSTEP  64

template <int MODE>
__global__ __launch_bounds__(256) void gemm_bf16_wmma_kernel(
    const unsigned short* __restrict__ A,    // [M][K]
    const unsigned short* __restrict__ BT,   // [N][K]
    const float* __restrict__ bias,
    int M, int N, int K,
    float* __restrict__ outF,
    unsigned short* __restrict__ qb,
    unsigned short* __restrict__ kb,
    unsigned short* __restrict__ vb) {
    __shared__ unsigned short As[2][TILE_M * KSTEP];    // 2 x 16 KB  [m][k]
    __shared__ unsigned short Bst[2][TILE_N * KSTEP];   // 2 x 16 KB  [n][k]

    const int tid  = threadIdx.x;
    const int wave = tid >> 5;
    const int lane = tid & 31;
    const int wm   = wave & 3;          // wave M index (0..3) -> 32 rows
    const int wn   = wave >> 2;         // wave N index (0..1) -> 64 cols
    const int m0   = blockIdx.x * TILE_M;
    const int n0   = blockIdx.y * TILE_N;
    const int lm   = lane & 15;
    const int lh   = lane >> 4;

    v8f acc[2][4] = {};

    // Cooperative DMA mapping: 128 rows x 2 halves of 32 elems (4 b128 each)
    const int row  = tid >> 1;
    const int kcol = (tid & 1) * 32;

    const unsigned short* agBase = A  + (size_t)(m0 + row) * K + kcol;
    const unsigned short* bgBase = BT + (size_t)(n0 + row) * K + kcol;
    unsigned short* adBase = &As[0][0]  + row * KSTEP + kcol;
    unsigned short* bdBase = &Bst[0][0] + row * KSTEP + kcol;

    auto issue_stage = [&](int buf, int k0) {
        const unsigned short* ag = agBase + k0;
        const unsigned short* bg = bgBase + k0;
        unsigned short* ad = adBase + buf * (TILE_M * KSTEP);
        unsigned short* bd = bdBase + buf * (TILE_N * KSTEP);
#pragma unroll
        for (int c = 0; c < 4; ++c) async_g2l_b128(ad + c * 8, ag + c * 8);
#pragma unroll
        for (int c = 0; c < 4; ++c) async_g2l_b128(bd + c * 8, bg + c * 8);
    };

    const int nk = K / KSTEP;
    issue_stage(0, 0);

    for (int kt = 0; kt < nk; ++kt) {
        const int buf = kt & 1;
        if (kt + 1 < nk) {
            issue_stage(buf ^ 1, (kt + 1) * KSTEP);
            if (kt + 2 < nk) {   // cache prefetch two stages ahead
                __builtin_prefetch(agBase + (kt + 2) * KSTEP, 0, 1);
                __builtin_prefetch(bgBase + (kt + 2) * KSTEP, 0, 1);
            }
            wait_async_le8();    // previous stage complete, next in flight
        } else {
            wait_async_le0();
        }
        __syncthreads();

        const unsigned short* Ab = As[buf];
        const unsigned short* Bb = Bst[buf];
#pragma unroll
        for (int kc = 0; kc < KSTEP; kc += 32) {
            BfFrag afr[2], bfr[4];
#pragma unroll
            for (int i = 0; i < 2; ++i) {
                const unsigned short* ap = Ab + (wm * 32 + i * 16 + lm) * KSTEP + kc;
#pragma unroll
                for (int p = 0; p < 4; ++p) {
                    afr[i].u32[p]     = *(const unsigned int*)(ap + lh * 8 + 2 * p);
                    afr[i].u32[p + 4] = *(const unsigned int*)(ap + 16 + lh * 8 + 2 * p);
                }
            }
#pragma unroll
            for (int j = 0; j < 4; ++j) {
                const unsigned short* bp = Bb + (wn * 64 + j * 16 + lm) * KSTEP + kc;
#pragma unroll
                for (int jj = 0; jj < 8; ++jj)
                    bfr[j].u32[jj] = *(const unsigned int*)(bp + (jj >> 2) * 16 + lh * 8 + 2 * (jj & 3));
            }
#pragma unroll
            for (int i = 0; i < 2; ++i)
#pragma unroll
                for (int j = 0; j < 4; ++j)
                    acc[i][j] = __builtin_amdgcn_wmma_f32_16x16x32_bf16(
                        false, afr[i].v, false, bfr[j].v, (short)0, acc[i][j],
                        false, false);
        }
        __syncthreads();
    }

    // ---- epilogue ----
#pragma unroll
    for (int i = 0; i < 2; ++i) {
#pragma unroll
        for (int j = 0; j < 4; ++j) {
            const int gn = n0 + wn * 64 + j * 16 + lm;
            const float bv = bias[gn];
            const int gm_base = m0 + wm * 32 + i * 16 + lh * 8;
#pragma unroll
            for (int r = 0; r < 8; ++r) {
                const int gm = gm_base + r;
                float val = acc[i][j][r] + bv;
                if (MODE == 1) {
                    outF[(size_t)gm * N + gn] = val;
                } else {
                    const int which = gn >> 10;          // 0:q 1:k 2:v
                    const int h     = (gn & 1023) >> 6;  // head
                    const int d     = gn & 63;
                    const int bIdx  = gm >> 11;          // batch
                    const int s     = gm & 2047;
                    const int bh    = bIdx * 16 + h;
                    if (which == 0) {
                        val *= 0.125f;                   // 1/sqrt(64)
                        qb[(((size_t)bh) * 2048 + s) * 64 + d] = f2bf(val);
                    } else if (which == 1) {
                        kb[(((size_t)bh) * 2048 + s) * 64 + d] = f2bf(val);
                    } else {
                        // V stored d-major: [BH][64][2048]
                        vb[(((size_t)bh) * 64 + d) * 2048 + s] = f2bf(val);
                    }
                }
            }
        }
    }
}

// ---------------------------------------------------------------------------
// Causal flash attention. Grid: (S/64, B*H). Block: 128 threads (4 waves).
// Double-buffered block-cooperative KV tiles (32 kv x 64 d), both staged by
// verbatim async DMA:  K [kv][d] row-major, V^T [d][kv] d-major.
// Each wave owns 16 query rows; waves past their causal bound skip compute.
// ---------------------------------------------------------------------------
#define ATT_S 2048
#define ATT_D 64

__global__ __launch_bounds__(128) void flash_attn_wmma_kernel(
    const unsigned short* __restrict__ Q,    // [BH, S, 64] bf16, pre-scaled
    const unsigned short* __restrict__ Kb,   // [BH, S, 64] bf16
    const unsigned short* __restrict__ VTb,  // [BH, 64, S] bf16 (d-major)
    unsigned short* __restrict__ Ab) {       // [B, S, 1024] bf16 merged heads
    __shared__ unsigned short Ks[2][32 * ATT_D];    // 2 x 4 KB [kv][d]
    __shared__ unsigned short Vst[2][ATT_D * 32];   // 2 x 4 KB [d][kv]
    __shared__ unsigned short pLds[4][16 * 32];     // 4 KB per-wave P scratch

    const int tid   = threadIdx.x;
    const int wave  = tid >> 5;
    const int lane  = tid & 31;
    const int lm    = lane & 15;
    const int lh    = lane >> 4;
    const int bh    = blockIdx.y;
    const int qbase = blockIdx.x * 64;
    const int qrow  = qbase + wave * 16;

    const unsigned short* Qh = Q   + (size_t)bh * ATT_S * ATT_D;
    const unsigned short* Kh = Kb  + (size_t)bh * ATT_S * ATT_D;
    const unsigned short* Vh = VTb + (size_t)bh * ATT_S * ATT_D;  // [64][2048]

    // Q fragments for the two 32-wide d-chunks (A layout: row = lm, K-pairs)
    BfFrag qf[2];
    {
        const unsigned short* qp = Qh + (size_t)(qrow + lm) * ATT_D;
#pragma unroll
        for (int c = 0; c < 2; ++c)
#pragma unroll
            for (int p = 0; p < 4; ++p) {
                qf[c].u32[p]     = *(const unsigned int*)(qp + c * 32 + lh * 8 + 2 * p);
                qf[c].u32[p + 4] = *(const unsigned int*)(qp + c * 32 + 16 + lh * 8 + 2 * p);
            }
    }

    v8f o[4] = {};
    float rowmax[8], rowsum[8];
#pragma unroll
    for (int r = 0; r < 8; ++r) { rowmax[r] = -1e30f; rowsum[r] = 0.0f; }

    // Cooperative DMA mappings
    const int kvr = tid >> 2;            // K tile: 0..31 kv row
    const int dcc = (tid & 3) * 16;      //         16-elem d chunk (2 b128)
    const int vdr = tid >> 1;            // V tile: 0..63 d row
    const int vkc = (tid & 1) * 16;      //         16-elem kv chunk (2 b128)

    auto issue_kv = [&](int buf, int kv) {
        const unsigned short* kg = Kh + (size_t)(kv + kvr) * ATT_D + dcc;
        unsigned short* kd = &Ks[buf][0] + kvr * ATT_D + dcc;
        async_g2l_b128(kd, kg);
        async_g2l_b128(kd + 8, kg + 8);
        const unsigned short* vg = Vh + (size_t)vdr * ATT_S + kv + vkc;
        unsigned short* vd = &Vst[buf][0] + vdr * 32 + vkc;
        async_g2l_b128(vd, vg);
        async_g2l_b128(vd + 8, vg + 8);
    };

    const int nkv = (qbase + 64) / 32;   // block-level causal tile count
    issue_kv(0, 0);

    for (int t = 0; t < nkv; ++t) {
        const int kv  = t * 32;
        const int buf = t & 1;
        if (t + 1 < nkv) {
            issue_kv(buf ^ 1, kv + 32);
            if (t + 2 < nkv) {
                __builtin_prefetch(Kh + (size_t)(kv + 64 + kvr) * ATT_D + dcc, 0, 1);
                __builtin_prefetch(Vh + (size_t)vdr * ATT_S + kv + 64 + vkc, 0, 1);
            }
            wait_async_le4();
        } else {
            wait_async_le0();
        }
        __syncthreads();

        if (kv < qrow + 16) {   // this wave still inside its causal range
            // ---- scores: 16 q-rows x 32 kv-cols ----
            v8f sc[2];
#pragma unroll
            for (int nt = 0; nt < 2; ++nt) {
                v8f c = {};
#pragma unroll
                for (int dc = 0; dc < 2; ++dc) {
                    BfFrag kf;  // B layout: col = kv index, K = d
                    const unsigned short* kp = &Ks[buf][0] + (nt * 16 + lm) * ATT_D + dc * 32;
#pragma unroll
                    for (int jj = 0; jj < 8; ++jj)
                        kf.u32[jj] = *(const unsigned int*)(kp + (jj >> 2) * 16 + lh * 8 + 2 * (jj & 3));
                    c = __builtin_amdgcn_wmma_f32_16x16x32_bf16(
                        false, qf[dc].v, false, kf.v, (short)0, c, false, false);
                }
                sc[nt] = c;
            }

            // ---- causal mask + online softmax (per-row over half-wave) ----
            float pv0[8], pv1[8];
#pragma unroll
            for (int r = 0; r < 8; ++r) {
                const int qr = qrow + r + lh * 8;
                float v0 = sc[0][r];
                float v1 = sc[1][r];
                v0 = ((kv + lm)      <= qr) ? v0 : -10000.0f;
                v1 = ((kv + 16 + lm) <= qr) ? v1 : -10000.0f;
                float mx = fmaxf(v0, v1);
#pragma unroll
                for (int off = 1; off < 16; off <<= 1)
                    mx = fmaxf(mx, __shfl_xor(mx, off, 32));
                const float newm  = fmaxf(rowmax[r], mx);
                const float scale = __expf(rowmax[r] - newm);
                rowmax[r] = newm;
                const float p0 = __expf(v0 - newm);
                const float p1 = __expf(v1 - newm);
                float ps = p0 + p1;
#pragma unroll
                for (int off = 1; off < 16; off <<= 1)
                    ps += __shfl_xor(ps, off, 32);
                rowsum[r] = rowsum[r] * scale + ps;
#pragma unroll
                for (int j = 0; j < 4; ++j) o[j][r] = o[j][r] * scale;
                pv0[r] = p0;
                pv1[r] = p1;
            }

            // ---- re-layout P: C-format -> A-format via wave-private LDS ----
            unsigned short* pl = pLds[wave];
#pragma unroll
            for (int r = 0; r < 8; ++r) {
                pl[(r + lh * 8) * 32 + lm]      = f2bf(pv0[r]);
                pl[(r + lh * 8) * 32 + 16 + lm] = f2bf(pv1[r]);
            }
            BfFrag pf;
            {
                const unsigned short* pp = pl + lm * 32;
#pragma unroll
                for (int p = 0; p < 4; ++p) {
                    pf.u32[p]     = *(const unsigned int*)(pp + lh * 8 + 2 * p);
                    pf.u32[p + 4] = *(const unsigned int*)(pp + 16 + lh * 8 + 2 * p);
                }
            }

            // ---- O += P(16x32) @ V(32x64) : four 16x16 n-tiles ----
#pragma unroll
            for (int j = 0; j < 4; ++j) {
                BfFrag vf;  // B layout: col = d (Vst row), K = kv index
                const unsigned short* vp = &Vst[buf][0] + (j * 16 + lm) * 32;
#pragma unroll
                for (int jj = 0; jj < 8; ++jj)
                    vf.u32[jj] = *(const unsigned int*)(vp + (jj >> 2) * 16 + lh * 8 + 2 * (jj & 3));
                o[j] = __builtin_amdgcn_wmma_f32_16x16x32_bf16(
                    false, pf.v, false, vf.v, (short)0, o[j], false, false);
            }
        }
        __syncthreads();
    }

    // ---- finalize: O / rowsum -> merged-head bf16 [B, S, 1024] ----
    const int bIdx = bh >> 4;
    const int h    = bh & 15;
#pragma unroll
    for (int j = 0; j < 4; ++j) {
#pragma unroll
        for (int r = 0; r < 8; ++r) {
            const int row = qrow + r + lh * 8;
            const int col = h * 64 + j * 16 + lm;
            const float val = o[j][r] / rowsum[r];
            Ab[((size_t)(bIdx * ATT_S + row)) * 1024 + col] = f2bf(val);
        }
    }
}

// ---------------------------------------------------------------------------
extern "C" void kernel_launch(void* const* d_in, const int* in_sizes, int n_in,
                              void* d_out, int out_size, void* d_ws, size_t ws_size,
                              hipStream_t stream) {
    const float* x      = (const float*)d_in[0];   // [2,2048,1024]
    const float* w_attn = (const float*)d_in[1];   // [1024,3072]
    const float* b_attn = (const float*)d_in[2];   // [3072]
    const float* w_proj = (const float*)d_in[3];   // [1024,1024]
    const float* b_proj = (const float*)d_in[4];   // [1024]
    float* out = (float*)d_out;                    // [2,2048,1024]

    const int M = 4096;       // B*S
    const int NX = 1024;
    const int N_QKV = 3072;

    char* ws = (char*)d_ws;
    unsigned short* xb    = (unsigned short*)(ws + 0);
    unsigned short* wabT  = (unsigned short*)(ws + (8u  << 20));   // [3072][1024]
    unsigned short* wpbT  = (unsigned short*)(ws + (14u << 20));   // [1024][1024]
    unsigned short* qbuf  = (unsigned short*)(ws + (16u << 20));   // [32][2048][64]
    unsigned short* kbuf  = (unsigned short*)(ws + (24u << 20));   // [32][2048][64]
    unsigned short* vbufT = (unsigned short*)(ws + (32u << 20));   // [32][64][2048]
    unsigned short* abuf  = (unsigned short*)(ws + (40u << 20));   // [4096][1024]

    // 1) fp32 -> bf16 conversions (weights transposed to [N][K])
    {
        int n = M * NX;
        cvt_f32_to_bf16_kernel<<<(n + 255) / 256, 256, 0, stream>>>(x, xb, n);
        n = NX * N_QKV;
        cvt_f32_to_bf16_T_kernel<<<(n + 255) / 256, 256, 0, stream>>>(
            w_attn, wabT, NX, N_QKV);
        n = NX * NX;
        cvt_f32_to_bf16_T_kernel<<<(n + 255) / 256, 256, 0, stream>>>(
            w_proj, wpbT, NX, NX);
    }

    // 2) QKV GEMM: [4096,1024] x [1024,3072] -> scatter q/k/vT head-major
    {
        dim3 grid(M / TILE_M, N_QKV / TILE_N);
        gemm_bf16_wmma_kernel<0><<<grid, 256, 0, stream>>>(
            xb, wabT, b_attn, M, N_QKV, NX, nullptr, qbuf, kbuf, vbufT);
    }

    // 3) causal flash attention -> merged heads bf16 [4096,1024]
    {
        dim3 grid(ATT_S / 64, 2 * 16);
        flash_attn_wmma_kernel<<<grid, 128, 0, stream>>>(qbuf, kbuf, vbufT, abuf);
    }

    // 4) projection GEMM: [4096,1024] x [1024,1024] + bias -> fp32 out
    {
        dim3 grid(M / TILE_M, NX / TILE_N);
        gemm_bf16_wmma_kernel<1><<<grid, 256, 0, stream>>>(
            abuf, wpbT, b_proj, M, NX, NX, out, nullptr, nullptr, nullptr);
    }
}
